// DenseSAKELayer_4681514353090
// MI455X (gfx1250) — matrix operand order repaired and checked by reading, not compiled, not run on hardware
//
#include <hip/hip_runtime.h>
#include <math.h>

#define NNODES 512
#define FD     32
#define HIDD   32
#define NHEADS 4
#define KRBF   50
#define NCH    128   // HIDD*NHEADS
#define NB     2
#define CUTV   5.0f
#define EPSV   1e-5f
#define PIF    3.14159265358979323846f

#define XOFF   (NB*NNODES*FD)                 // 32768
#define VOFF   (NB*NNODES*FD + NB*NNODES*3)   // 35840

typedef __attribute__((ext_vector_type(16))) _Float16 v16h;
typedef __attribute__((ext_vector_type(8)))  float    v8f;

__device__ __forceinline__ v8f wmma16(v16h a, v16h b, v8f c) {
  // D = A(16x32 f16) * B(32x16 f16) + C(16x16 f32)
  return __builtin_amdgcn_wmma_f32_16x16x32_f16(false, a, false, b, (short)0, c, false, false);
}

__device__ __forceinline__ float siluf(float xx) { return xx / (1.0f + __expf(-xx)); }

__device__ __forceinline__ v16h ld_v16h(const _Float16* p) {
  v16h r;
#pragma unroll
  for (int e = 0; e < 16; ++e) r[e] = p[e];
  return r;
}

// element e of the A-operand v16h -> K index within a 32-wide K slab
__device__ __forceinline__ int a_kmap(int e, int laneHi) {
  return e + (laneHi ? 16 : 0);
}

// ---------------------------------------------------------------------------
// Prep: h -> f16, weights -> WMMA B-operand lane layout
// B layout: tile = [kslab][ntile][lane][16 halfs]; col n = base + (lane&15),
//           K = kslab*32 + e + ((lane&16)?16:0)
// ---------------------------------------------------------------------------
__global__ void sake_prep(const float* __restrict__ h, const float* __restrict__ W_in,
                          const float* __restrict__ W_o1, const float* __restrict__ W_o2,
                          _Float16* __restrict__ h16, _Float16* __restrict__ winT,
                          _Float16* __restrict__ wo1T, _Float16* __restrict__ wo2T) {
  int tid = blockIdx.x * blockDim.x + threadIdx.x;
  int stride = gridDim.x * blockDim.x;
  for (int idx = tid; idx < NB * NNODES * FD; idx += stride)
    h16[idx] = (_Float16)h[idx];
  for (int idx = tid; idx < 2 * 4 * 32 * 16; idx += stride) {   // W_in: K=64 (2 slabs), N=50->64 (4 tiles)
    int e = idx & 15, lane = (idx >> 4) & 31, nt = (idx >> 9) & 3, s = idx >> 11;
    int k = s * 32 + e + ((lane & 16) ? 16 : 0);
    int n = nt * 16 + (lane & 15);
    winT[idx] = (_Float16)((n < KRBF) ? W_in[k * KRBF + n] : 0.0f);
  }
  for (int idx = tid; idx < 4 * 2 * 32 * 16; idx += stride) {   // W_o1: K=115->128 (4 slabs), N=32 (2 tiles)
    int e = idx & 15, lane = (idx >> 4) & 31, nt = (idx >> 9) & 1, s = idx >> 10;
    int k = s * 32 + e + ((lane & 16) ? 16 : 0);
    int n = nt * 16 + (lane & 15);
    wo1T[idx] = (_Float16)((k < 115) ? W_o1[k * 32 + n] : 0.0f);
  }
  for (int idx = tid; idx < 2 * 32 * 16; idx += stride) {       // W_o2: K=32, N=32 (2 tiles)
    int e = idx & 15, lane = (idx >> 4) & 31, nt = (idx >> 9) & 1;
    int k = e + ((lane & 16) ? 16 : 0);
    int n = nt * 16 + (lane & 15);
    wo2T[idx] = (_Float16)W_o2[k * 32 + n];
  }
}

__device__ __forceinline__ void block_reduce4(float* red, float* vv, int t, bool domax) {
  red[t * 4 + 0] = vv[0]; red[t * 4 + 1] = vv[1];
  red[t * 4 + 2] = vv[2]; red[t * 4 + 3] = vv[3];
  __syncthreads();
  for (int s = 128; s > 0; s >>= 1) {
    if (t < s) {
#pragma unroll
      for (int hh = 0; hh < 4; ++hh) {
        float a = red[t * 4 + hh], b = red[(t + s) * 4 + hh];
        red[t * 4 + hh] = domax ? fmaxf(a, b) : (a + b);
      }
    }
    __syncthreads();
  }
  vv[0] = red[0]; vv[1] = red[1]; vv[2] = red[2]; vv[3] = red[3];
  __syncthreads();
}

// ---------------------------------------------------------------------------
// Main fused kernel: one block per (b, i) row
// ---------------------------------------------------------------------------
__global__ __launch_bounds__(256)
void sake_main(const float* __restrict__ x, const float* __restrict__ v,
               const float* __restrict__ h,
               const float* __restrict__ means, const float* __restrict__ betas,
               const float* __restrict__ b_in,
               const float* __restrict__ b_o1, const float* __restrict__ b_o2,
               const float* __restrict__ W_sem, const float* __restrict__ b_sem,
               const float* __restrict__ W_pn1, const float* __restrict__ b_pn1,
               const float* __restrict__ W_pn2, const float* __restrict__ b_pn2,
               const float* __restrict__ W_n1, const float* __restrict__ b_n1,
               const float* __restrict__ W_n2, const float* __restrict__ b_n2,
               const float* __restrict__ W_v1, const float* __restrict__ b_v1,
               const float* __restrict__ W_v2, const float* __restrict__ W_vm,
               const float* __restrict__ log_gamma,
               const _Float16* __restrict__ h16g, const _Float16* __restrict__ winTg,
               const _Float16* __restrict__ wo1Tg, const _Float16* __restrict__ wo2Tg,
               float* __restrict__ outp) {
  __shared__ __attribute__((aligned(16))) _Float16 s_h16[NNODES * FD];     // 32 KB
  __shared__ __attribute__((aligned(16))) _Float16 s_he[NNODES * HIDD];    // 32 KB
  __shared__ __attribute__((aligned(16))) _Float16 s_winT[2 * 4 * 32 * 16];// 8 KB
  __shared__ __attribute__((aligned(16))) _Float16 s_wo1T[4 * 2 * 32 * 16];// 8 KB
  __shared__ __attribute__((aligned(16))) _Float16 s_wo2T[2 * 32 * 16];    // 2 KB
  __shared__ _Float16 s_stage1[8 * 16 * 64];                               // 16 KB
  __shared__ _Float16 s_stage2[8 * 16 * 32];                               // 8 KB
  __shared__ float s_sem[NNODES * NHEADS];                                 // 8 KB
  __shared__ float s_eucl[NNODES * NHEADS];                                // 8 KB
  __shared__ float s_norm[NNODES], s_ea[NNODES], s_cw[NNODES];             // 6 KB
  __shared__ float s_unit[NNODES * 3];                                     // 6 KB
  __shared__ float s_red[256 * 4];                                         // 4 KB
  __shared__ float s_means[KRBF], s_betas[KRBF], s_bin[KRBF];
  __shared__ float s_bo1[HIDD], s_bo2[HIDD];
  __shared__ float s_wsem[HIDD * NHEADS], s_bsem[NHEADS], s_gamma[NHEADS];
  __shared__ _Float16 s_hi16[FD];
  __shared__ float s_hi[FD], s_xi[3], s_vi[3];
  __shared__ float s_hesum[NCH], s_csum[NCH * 3], s_hcin[NCH];
  __shared__ float s_z1[HIDD], s_hcomb[HIDD], s_y1[HIDD], s_hn[FD], s_g1[HIDD];

  const int t = threadIdx.x;
  const int bi = blockIdx.x;
  const int b = bi >> 9;
  const int i = bi & (NNODES - 1);
  const int node = b * NNODES + i;

  // ------------------------- Phase 0: stage inputs -------------------------
  {
    const uint4* gh = (const uint4*)(h16g + b * NNODES * FD);
    for (int idx = t; idx < NNODES * FD / 8; idx += 256) ((uint4*)s_h16)[idx] = gh[idx];
    for (int idx = t; idx < 2 * 4 * 32 * 16 / 8; idx += 256) ((uint4*)s_winT)[idx] = ((const uint4*)winTg)[idx];
    for (int idx = t; idx < 4 * 2 * 32 * 16 / 8; idx += 256) ((uint4*)s_wo1T)[idx] = ((const uint4*)wo1Tg)[idx];
    for (int idx = t; idx < 2 * 32 * 16 / 8; idx += 256) ((uint4*)s_wo2T)[idx] = ((const uint4*)wo2Tg)[idx];
    if (t < KRBF) { s_means[t] = means[t]; s_betas[t] = betas[t]; s_bin[t] = b_in[t]; }
    if (t >= 64 && t < 96) {
      int c = t - 64;
      s_bo1[c] = b_o1[c]; s_bo2[c] = b_o2[c];
      float hv = h[node * FD + c];
      s_hi[c] = hv; s_hi16[c] = (_Float16)hv;
    }
    if (t >= 96 && t < 224) s_wsem[t - 96] = W_sem[t - 96];
    if (t >= 224 && t < 228) {
      s_bsem[t - 224] = b_sem[t - 224];
      s_gamma[t - 224] = __expf(log_gamma[t - 224]);
    }
    if (t >= 228 && t < 231) {
      s_xi[t - 228] = x[node * 3 + (t - 228)];
      s_vi[t - 228] = v[node * 3 + (t - 228)];
    }
    float xi0 = x[node * 3 + 0], xi1 = x[node * 3 + 1], xi2 = x[node * 3 + 2];
    for (int jj = t; jj < NNODES; jj += 256) {
      float dx0 = x[(b * NNODES + jj) * 3 + 0] - xi0;
      float dx1 = x[(b * NNODES + jj) * 3 + 1] - xi1;
      float dx2 = x[(b * NNODES + jj) * 3 + 2] - xi2;
      float n2 = dx0 * dx0 + dx1 * dx1 + dx2 * dx2;
      float nr = sqrtf(fmaxf(n2, 0.0f) + EPSV);
      s_norm[jj] = nr;
      s_ea[jj] = __expf(-nr);                       // alpha = 5/CUT = 1
      s_cw[jj] = (nr < CUTV) ? 0.5f * (cosf(nr * PIF / CUTV) + 1.0f) : 0.0f;
      float inv = 1.0f / (nr + EPSV);
      s_unit[jj * 3 + 0] = dx0 * inv;
      s_unit[jj * 3 + 1] = dx1 * inv;
      s_unit[jj * 3 + 2] = dx2 * inv;
    }
  }
  __syncthreads();

  // ------------------- Phase 1: WMMA edge-MLP over j tiles -----------------
  {
    const int lane = t & 31, wave = t >> 5;
    const int laneHi = (lane >> 4) & 1, lcol = lane & 15;
    _Float16* stg1 = &s_stage1[wave * 16 * 64];
    _Float16* stg2 = &s_stage2[wave * 16 * 32];

    v16h a_hi;                                      // h_i broadcast (K slab = h_cat[32..63])
#pragma unroll
    for (int e = 0; e < 16; ++e) a_hi[e] = s_hi16[a_kmap(e, laneHi)];

    for (int jt = wave; jt < NNODES / 16; jt += 8) {
      const int jbase = jt * 16;
      const int jrow = jbase + lcol;

      v16h a_hj;                                    // h_j rows (K slab = h_cat[0..31])
#pragma unroll
      for (int e = 0; e < 16; ++e) a_hj[e] = s_h16[jrow * FD + a_kmap(e, laneHi)];

      // ---- GEMM1: h_cat(16x64) @ W_in -> hk(16x50), then *= smearing ----
#pragma unroll
      for (int nt = 0; nt < 4; ++nt) {
        v8f acc = {};
        acc = wmma16(a_hj, ld_v16h(&s_winT[((0 * 4 + nt) * 32 + lane) * 16]), acc);
        acc = wmma16(a_hi, ld_v16h(&s_winT[((1 * 4 + nt) * 32 + lane) * 16]), acc);
#pragma unroll
        for (int r = 0; r < 8; ++r) {
          int m = r + laneHi * 8;
          int jc = jbase + m;
          int k = nt * 16 + lcol;
          float val = 0.0f;
          if (k < KRBF) {
            float d = s_ea[jc] - s_means[k];
            float sm = s_cw[jc] * __expf(-s_betas[k] * d * d);
            val = (acc[r] + s_bin[k]) * sm;
          }
          stg1[m * 64 + k] = (_Float16)val;
        }
      }

      // ---- A operands for concat K-slabs 2,3 (from staging transpose) ----
      v16h a2, a3;
#pragma unroll
      for (int e = 0; e < 16; ++e) {
        int kp = a_kmap(e, laneHi);
        a2[e] = stg1[lcol * 64 + kp];               // _x[0..31]
        a3[e] = (kp <= 17) ? stg1[lcol * 64 + 32 + kp]
              : (kp == 18) ? (_Float16)s_norm[jrow] // K=114 -> norm feature
                           : (_Float16)0.0f;
      }

      // ---- GEMM2: concat(16x128) @ W_o1 -> silu -> (16x32) ----
#pragma unroll
      for (int nt = 0; nt < 2; ++nt) {
        v8f acc = {};
        acc = wmma16(a_hj, ld_v16h(&s_wo1T[((0 * 2 + nt) * 32 + lane) * 16]), acc);
        acc = wmma16(a_hi, ld_v16h(&s_wo1T[((1 * 2 + nt) * 32 + lane) * 16]), acc);
        acc = wmma16(a2,   ld_v16h(&s_wo1T[((2 * 2 + nt) * 32 + lane) * 16]), acc);
        acc = wmma16(a3,   ld_v16h(&s_wo1T[((3 * 2 + nt) * 32 + lane) * 16]), acc);
#pragma unroll
        for (int r = 0; r < 8; ++r) {
          int m = r + laneHi * 8;
          int n = nt * 16 + lcol;
          float y = acc[r] + s_bo1[n];
          stg2[m * 32 + n] = (_Float16)siluf(y);
        }
      }

      // ---- GEMM3: (16x32) @ W_o2 -> h_e tile ----
      v16h a4;
#pragma unroll
      for (int e = 0; e < 16; ++e) a4[e] = stg2[lcol * 32 + a_kmap(e, laneHi)];
#pragma unroll
      for (int nt = 0; nt < 2; ++nt) {
        v8f acc = {};
        acc = wmma16(a4, ld_v16h(&s_wo2T[(nt * 32 + lane) * 16]), acc);
#pragma unroll
        for (int r = 0; r < 8; ++r) {
          int m = r + laneHi * 8;
          int c = nt * 16 + lcol;
          s_he[(jbase + m) * HIDD + c] = (_Float16)(acc[r] + s_bo2[c]);
        }
      }
    }
  }
  __syncthreads();

  // ----------------- Phase 2: sem logits + 3 softmaxes over j --------------
  for (int jj = t; jj < NNODES; jj += 256) {
    const _Float16* her = &s_he[jj * HIDD];
#pragma unroll
    for (int hh = 0; hh < NHEADS; ++hh) {
      float acc = s_bsem[hh];
      for (int c = 0; c < HIDD; ++c) acc += (float)her[c] * s_wsem[c * NHEADS + hh];
      float l = (acc > 0.0f) ? acc : 0.2f * acc;    // leaky_relu(0.2)
      if (jj == i) l -= 1e5f;                       // self mask
      s_sem[jj * NHEADS + hh] = l;
    }
  }
  __syncthreads();

  {
    // sem softmax (in place)
    float mx[4] = {-3.4e38f, -3.4e38f, -3.4e38f, -3.4e38f};
    for (int jj = t; jj < NNODES; jj += 256)
      for (int hh = 0; hh < 4; ++hh) mx[hh] = fmaxf(mx[hh], s_sem[jj * 4 + hh]);
    block_reduce4(s_red, mx, t, true);
    float sm[4] = {0, 0, 0, 0};
    for (int jj = t; jj < NNODES; jj += 256)
      for (int hh = 0; hh < 4; ++hh) {
        float e = __expf(s_sem[jj * 4 + hh] - mx[hh]);
        s_sem[jj * 4 + hh] = e; sm[hh] += e;
      }
    block_reduce4(s_red, sm, t, false);
    for (int jj = t; jj < NNODES; jj += 256)
      for (int hh = 0; hh < 4; ++hh) s_sem[jj * 4 + hh] *= 1.0f / sm[hh];

    // eucl softmax
    float mx2[4] = {-3.4e38f, -3.4e38f, -3.4e38f, -3.4e38f};
    for (int jj = t; jj < NNODES; jj += 256) {
      float base = s_norm[jj] + ((jj == i) ? 1e5f : 0.0f);
      for (int hh = 0; hh < 4; ++hh) {
        float l = -base * s_gamma[hh];
        s_eucl[jj * 4 + hh] = l;
        mx2[hh] = fmaxf(mx2[hh], l);
      }
    }
    block_reduce4(s_red, mx2, t, true);
    float sm2[4] = {0, 0, 0, 0};
    for (int jj = t; jj < NNODES; jj += 256)
      for (int hh = 0; hh < 4; ++hh) {
        float e = __expf(s_eucl[jj * 4 + hh] - mx2[hh]);
        s_eucl[jj * 4 + hh] = e; sm2[hh] += e;
      }
    block_reduce4(s_red, sm2, t, false);

    // p = eucl*sem, then comb = softmax(p) over j -> stored back into s_sem
    float mx3[4] = {-3.4e38f, -3.4e38f, -3.4e38f, -3.4e38f};
    for (int jj = t; jj < NNODES; jj += 256)
      for (int hh = 0; hh < 4; ++hh) {
        float p = (s_eucl[jj * 4 + hh] / sm2[hh]) * s_sem[jj * 4 + hh];
        s_eucl[jj * 4 + hh] = p;
        mx3[hh] = fmaxf(mx3[hh], p);
      }
    block_reduce4(s_red, mx3, t, true);
    float sm3[4] = {0, 0, 0, 0};
    for (int jj = t; jj < NNODES; jj += 256)
      for (int hh = 0; hh < 4; ++hh) {
        float e = __expf(s_eucl[jj * 4 + hh] - mx3[hh]);
        s_sem[jj * 4 + hh] = e; sm3[hh] += e;
      }
    block_reduce4(s_red, sm3, t, false);
    for (int jj = t; jj < NNODES; jj += 256)
      for (int hh = 0; hh < 4; ++hh) s_sem[jj * 4 + hh] *= 1.0f / sm3[hh];  // comb
  }
  __syncthreads();

  // --------- Phase 3: reductions over j (h_e_sum, comb_sum with unit) ------
  {
    int cc = t & 127, half = t >> 7;
    int c = cc >> 2, hh = cc & 3;
    float s0 = 0, s1 = 0, s2 = 0, s3 = 0;
    int j0 = half * 256;
    for (int jj = j0; jj < j0 + 256; ++jj) {
      float he = (float)s_he[jj * HIDD + c];
      float w = he * s_sem[jj * 4 + hh];            // h_e_att channel cc
      s0 += w;
      s1 += w * s_unit[jj * 3 + 0];
      s2 += w * s_unit[jj * 3 + 1];
      s3 += w * s_unit[jj * 3 + 2];
    }
    s_red[t * 4 + 0] = s0; s_red[t * 4 + 1] = s1;
    s_red[t * 4 + 2] = s2; s_red[t * 4 + 3] = s3;
  }
  __syncthreads();
  if (t < NCH) {
    s_hesum[t] = s_red[t * 4] + s_red[(t + 128) * 4];
    float c0 = (s_red[t * 4 + 1] + s_red[(t + 128) * 4 + 1]) * (1.0f / NNODES);
    float c1 = (s_red[t * 4 + 2] + s_red[(t + 128) * 4 + 2]) * (1.0f / NNODES);
    float c2 = (s_red[t * 4 + 3] + s_red[(t + 128) * 4 + 3]) * (1.0f / NNODES);
    s_csum[t * 3 + 0] = c0; s_csum[t * 3 + 1] = c1; s_csum[t * 3 + 2] = c2;
    s_hcin[t] = c0 * c0 + c1 * c1 + c2 * c2;
  }
  __syncthreads();

  // ----------------------- Phase 4: per-node MLPs --------------------------
  if (t < HIDD) {
    float u = b_pn1[t];
    for (int cc = 0; cc < NCH; ++cc) u += s_hcin[cc] * W_pn1[cc * HIDD + t];
    s_z1[t] = siluf(u);
  }
  __syncthreads();
  if (t < HIDD) {
    float u = b_pn2[t];
    for (int c = 0; c < HIDD; ++c) u += s_z1[c] * W_pn2[c * HIDD + t];
    s_hcomb[t] = siluf(u);
  }
  __syncthreads();
  if (t < HIDD) {
    float u = b_n1[t];
    for (int f = 0; f < FD; ++f)   u += s_hi[f] * W_n1[f * HIDD + t];
    for (int cc = 0; cc < NCH; ++cc) u += s_hesum[cc] * W_n1[(FD + cc) * HIDD + t];
    for (int c = 0; c < HIDD; ++c) u += s_hcomb[c] * W_n1[(FD + NCH + c) * HIDD + t];
    s_y1[t] = siluf(u);
  }
  __syncthreads();
  if (t < FD) {
    float u = b_n2[t];
    for (int c = 0; c < HIDD; ++c) u += s_y1[c] * W_n2[c * FD + t];
    float hn = s_hi[t] + siluf(u);
    s_hn[t] = hn;
    outp[node * FD + t] = hn;
  }
  __syncthreads();
  if (t < HIDD) {
    float u = b_v1[t];
    for (int c = 0; c < FD; ++c) u += s_hn[c] * W_v1[c * HIDD + t];
    s_g1[t] = siluf(u);
  }
  __syncthreads();
  if (t == 0) {
    float g = 0;
    for (int c = 0; c < HIDD; ++c) g += s_g1[c] * W_v2[c];
    s_red[0] = g;
  }
  __syncthreads();
  if (t < 3) {
    float dv = 0;
    for (int cc = 0; cc < NCH; ++cc) dv += s_csum[cc * 3 + t] * W_vm[cc];
    float vn = s_red[0] * s_vi[t] + dv;
    outp[XOFF + node * 3 + t] = s_xi[t] + vn;
    outp[VOFF + node * 3 + t] = vn;
  }
}

extern "C" void kernel_launch(void* const* d_in, const int* in_sizes, int n_in,
                              void* d_out, int out_size, void* d_ws, size_t ws_size,
                              hipStream_t stream) {
  (void)in_sizes; (void)n_in; (void)out_size; (void)ws_size;
  const float* h        = (const float*)d_in[0];
  const float* x        = (const float*)d_in[1];
  const float* v        = (const float*)d_in[2];
  const float* means    = (const float*)d_in[3];
  const float* betas    = (const float*)d_in[4];
  const float* W_in     = (const float*)d_in[5];
  const float* b_in     = (const float*)d_in[6];
  const float* W_o1     = (const float*)d_in[7];
  const float* b_o1     = (const float*)d_in[8];
  const float* W_o2     = (const float*)d_in[9];
  const float* b_o2     = (const float*)d_in[10];
  const float* W_sem    = (const float*)d_in[11];
  const float* b_sem    = (const float*)d_in[12];
  const float* W_pn1    = (const float*)d_in[13];
  const float* b_pn1    = (const float*)d_in[14];
  const float* W_pn2    = (const float*)d_in[15];
  const float* b_pn2    = (const float*)d_in[16];
  const float* W_n1     = (const float*)d_in[17];
  const float* b_n1     = (const float*)d_in[18];
  const float* W_n2     = (const float*)d_in[19];
  const float* b_n2     = (const float*)d_in[20];
  const float* W_v1     = (const float*)d_in[21];
  const float* b_v1     = (const float*)d_in[22];
  const float* W_v2     = (const float*)d_in[23];
  const float* W_vm     = (const float*)d_in[24];
  const float* log_gam  = (const float*)d_in[25];

  _Float16* h16  = (_Float16*)d_ws;                 // 32768 halfs
  _Float16* winT = h16 + NB * NNODES * FD;          // 4096 halfs
  _Float16* wo1T = winT + 2 * 4 * 32 * 16;          // 4096 halfs
  _Float16* wo2T = wo1T + 4 * 2 * 32 * 16;          // 1024 halfs

  sake_prep<<<64, 256, 0, stream>>>(h, W_in, W_o1, W_o2, h16, winT, wo1T, wo2T);
  sake_main<<<NB * NNODES, 256, 0, stream>>>(
      x, v, h, means, betas, b_in, b_o1, b_o2, W_sem, b_sem,
      W_pn1, b_pn1, W_pn2, b_pn2, W_n1, b_n1, W_n2, b_n2,
      W_v1, b_v1, W_v2, W_vm, log_gam,
      h16, winT, wo1T, wo2T, (float*)d_out);
}
